// CausalSelfAttention_75986561401052
// MI455X (gfx1250) — compile-verified, matrix operand order checked by reading, not run on hardware
//
#include <hip/hip_runtime.h>
#include <hip/hip_bf16.h>

// ---------------------------------------------------------------------------
// CDNA5 (gfx1250) causal self-attention, bf16 WMMA pipeline.
//   B=2, T=2048, C=1024, H=16, D=64
// ---------------------------------------------------------------------------

typedef __bf16 bf16;
typedef __attribute__((ext_vector_type(16))) __bf16 v16bf;
typedef __attribute__((ext_vector_type(8)))  __bf16 v8bf;
typedef __attribute__((ext_vector_type(8)))  float  v8f;

#define T_SEQ   2048
#define C_EMB   1024
#define N_HEAD  16
#define HEAD_D  64
#define M_ROWS  4096   // B*T

// 1/sqrt(64) * log2(e): softmax done in exp2 domain, scale folded into Q.
#define Q_SCALE 0.1803368801111204f

// ---- WMMA wrapper: D = A(16x32 bf16) * B(32x16 bf16) + C(16x16 f32) -------
__device__ __forceinline__ v8f wmma_bf16(v16bf a, v16bf b, v8f c) {
  return __builtin_amdgcn_wmma_f32_16x16x32_bf16(
      /*neg_a=*/false, a, /*neg_b=*/false, b,
      /*c_mod=*/(short)0, c, /*reuse_a=*/false, /*reuse_b=*/false);
}

// ---- Fragment loader (A and B share the mirrored per-lane layout) ---------
// A 16x32 (MxK): lane row r = lane&15, kb = 8*(lane>=16);
//   elems[0..7]  = row[kb + 0..7], elems[8..15] = row[kb + 16..23]
// B 32x16 (KxN): identical with r = N column, data contiguous along K per lane.
__device__ __forceinline__ v16bf load_frag(const bf16* p, int ld) {
  const int lane = threadIdx.x & 31;
  const int r    = lane & 15;
  const int kb   = (lane >> 4) * 8;
  const bf16* base = p + (size_t)r * ld + kb;
  v8bf lo = *(const v8bf*)(base);        // 16B aligned (ld multiple of 8)
  v8bf hi = *(const v8bf*)(base + 16);
  return __builtin_shufflevector(lo, hi, 0,1,2,3,4,5,6,7,8,9,10,11,12,13,14,15);
}

// ---- CDNA5 async copy: global -> LDS, 16B per call, ASYNCcnt-tracked ------
__device__ __forceinline__ void async_copy_b128(const void* gsrc, unsigned lds_off) {
  asm volatile("global_load_async_to_lds_b128 %0, %1, off"
               :: "v"(lds_off), "v"((unsigned long long)gsrc)
               : "memory");
}
__device__ __forceinline__ void wait_asynccnt0() {
  asm volatile("s_wait_asynccnt 0" ::: "memory");
}

// ---------------------------------------------------------------------------
// Kernel 1: fp32 -> bf16 convert
// ---------------------------------------------------------------------------
__global__ void __launch_bounds__(256) cvt_f32_bf16(const float* __restrict__ src,
                                                    bf16* __restrict__ dst, int n) {
  int i = blockIdx.x * 256 + threadIdx.x;
  if (i < n) dst[i] = (bf16)src[i];
}

// ---------------------------------------------------------------------------
// Kernel 2/4: GEMM  out[m][n] = sum_k A[m][k] * W[n][k]   (K = 1024 always)
// MODE 0: (A=x, W=Wq)  store Q   bf16 [(b*16+h)][t][d], scaled by Q_SCALE
// MODE 1: (A=x, W=Wk)  store K   bf16 [(b*16+h)][t][d]
// MODE 2: (A=Wv, W=x)  SWAPPED GEMM: out[c][tok] = V^T; coalesced row store
// MODE 3: (A=attn, W=Wo) store f32 row-major [m][n] -> d_out
// Block = 128 threads (4 waves, 2x2); wave tile 64(M)x32(N); block 128x64.
// ---------------------------------------------------------------------------
template <int MODE>
__device__ __forceinline__ void store_tile(const v8f& c, int gm0, int gn0,
                                           bf16* __restrict__ qkv,
                                           float* __restrict__ outp) {
  const int lane = threadIdx.x & 15;
  const int half = (threadIdx.x & 31) >> 4;
#pragma unroll
  for (int i = 0; i < 8; ++i) {
    const int gm = gm0 + i + 8 * half;     // row  (C/D layout: m = i + 8*half)
    const int gn = gn0 + lane;             // col  (n = lane&15)
    float v = c[i];
    if (MODE == 0 || MODE == 1) {
      const int b = gm >> 11, t = gm & (T_SEQ - 1);
      const int h = gn >> 6,  d = gn & (HEAD_D - 1);
      const float sv = (MODE == 0) ? v * Q_SCALE : v;
      qkv[(((size_t)(b * N_HEAD + h) * T_SEQ + t) << 6) + d] = (bf16)sv;
    } else if (MODE == 2) {
      // swapped: gm = channel c, gn = token index
      const int h = gm >> 6,  d = gm & (HEAD_D - 1);
      const int b = gn >> 11, t = gn & (T_SEQ - 1);
      qkv[(((size_t)(b * N_HEAD + h) * HEAD_D + d) << 11) + t] = (bf16)v;
    } else {
      outp[(size_t)gm * C_EMB + gn] = v;
    }
  }
}

template <int MODE>
__global__ void __launch_bounds__(128) gemm_xwT(const bf16* __restrict__ A,
                                                const bf16* __restrict__ W,
                                                bf16* __restrict__ qkv,
                                                float* __restrict__ outp) {
  const int wid = threadIdx.x >> 5;
  const int m0  = blockIdx.x * 128 + (wid & 1) * 64;
  const int n0  = blockIdx.y * 64  + (wid >> 1) * 32;

  v8f c00 = {}, c01 = {}, c10 = {}, c11 = {};
  v8f c20 = {}, c21 = {}, c30 = {}, c31 = {};
  for (int k = 0; k < C_EMB; k += 32) {
    v16bf a0 = load_frag(A + (size_t)(m0 +  0) * C_EMB + k, C_EMB);
    v16bf a1 = load_frag(A + (size_t)(m0 + 16) * C_EMB + k, C_EMB);
    v16bf a2 = load_frag(A + (size_t)(m0 + 32) * C_EMB + k, C_EMB);
    v16bf a3 = load_frag(A + (size_t)(m0 + 48) * C_EMB + k, C_EMB);
    v16bf b0 = load_frag(W + (size_t)(n0 +  0) * C_EMB + k, C_EMB);
    v16bf b1 = load_frag(W + (size_t)(n0 + 16) * C_EMB + k, C_EMB);
    c00 = wmma_bf16(a0, b0, c00);  c01 = wmma_bf16(a0, b1, c01);
    c10 = wmma_bf16(a1, b0, c10);  c11 = wmma_bf16(a1, b1, c11);
    c20 = wmma_bf16(a2, b0, c20);  c21 = wmma_bf16(a2, b1, c21);
    c30 = wmma_bf16(a3, b0, c30);  c31 = wmma_bf16(a3, b1, c31);
  }
  store_tile<MODE>(c00, m0,      n0,      qkv, outp);
  store_tile<MODE>(c01, m0,      n0 + 16, qkv, outp);
  store_tile<MODE>(c10, m0 + 16, n0,      qkv, outp);
  store_tile<MODE>(c11, m0 + 16, n0 + 16, qkv, outp);
  store_tile<MODE>(c20, m0 + 32, n0,      qkv, outp);
  store_tile<MODE>(c21, m0 + 32, n0 + 16, qkv, outp);
  store_tile<MODE>(c30, m0 + 48, n0,      qkv, outp);
  store_tile<MODE>(c31, m0 + 48, n0 + 16, qkv, outp);
}

// ---------------------------------------------------------------------------
// Kernel 3: flash attention.
// Block = 128 threads (4 waves), one (b,h) per blockIdx.y, 64 query rows per
// block (16 per wave). K/V chunks staged into LDS once per block via
// GLOBAL_LOAD_ASYNC_TO_LDS_B128 (double-buffered, one s_wait_asynccnt + one
// barrier per iteration). All 8 fragment loads are hoisted ahead of the WMMAs
// so ds_loads clause up and V-frag latency hides under softmax. Causal mask
// applied only on partial chunks (wave-uniform branch).
// ---------------------------------------------------------------------------
#define KSTRIDE 72   // 64 + 8 bf16  (144B rows: 9r mod 16 bank rotation)
#define VSTRIDE 40   // 32 + 8 bf16  ( 80B rows: 5r mod 16 bank rotation)
#define SSTRIDE 36   // 32 + 4 f32

__global__ void __launch_bounds__(128) flash_attn(const bf16* __restrict__ Q,
                                                  const bf16* __restrict__ K,
                                                  const bf16* __restrict__ VT,
                                                  bf16* __restrict__ Aout) {
  __shared__ __align__(16) bf16  lds_k[2][32 * KSTRIDE];   // 2 x 4.5 KB
  __shared__ __align__(16) bf16  lds_v[2][64 * VSTRIDE];   // 2 x 5.0 KB
  __shared__ __align__(16) float lds_s[4][16 * SSTRIDE];   // 4 x 2.25 KB

  const int tid  = threadIdx.x;
  const int wid  = tid >> 5;
  const int lane = tid & 31;
  const int n    = lane & 15;          // C-layout column / A-layout row
  const int half = lane >> 4;
  const int kb   = half * 8;           // A-layout k-base for this lane
  const int q0b  = blockIdx.x * 64;    // block's query base
  const int q0   = q0b + wid * 16;     // this wave's query tile base
  const int bh   = blockIdx.y;         // b*16 + h

  const bf16* Qh  = Q  + (size_t)bh * T_SEQ * HEAD_D;
  const bf16* Kh  = K  + (size_t)bh * T_SEQ * HEAD_D;
  const bf16* VTh = VT + (size_t)bh * HEAD_D * T_SEQ;
  float* ls = lds_s[wid];

  // Q fragments (K-dim = head dim, two 32-wide chunks)
  const v16bf aq0 = load_frag(Qh + (size_t)q0 * HEAD_D + 0,  HEAD_D);
  const v16bf aq1 = load_frag(Qh + (size_t)q0 * HEAD_D + 32, HEAD_D);

  v8f acc0 = {}, acc1 = {}, acc2 = {}, acc3 = {};
  float m_run = -1e30f;   // stats for row (lane&15), duplicated in both k-halves
  float l_run = 0.0f;

  // cooperative async staging of one 32-key chunk into buffer `buf`
  auto stage = [&](int c, int buf) {
    const int kt0 = c * 32;
    const unsigned kbase = (unsigned)(unsigned long long)&lds_k[buf][0];
    const unsigned vbase = (unsigned)(unsigned long long)&lds_v[buf][0];
    // K chunk: 32 rows x 128B; thread -> (row, row+16) x one 16B segment
    const int kr = tid >> 3, ks = tid & 7;
    async_copy_b128(Kh + (size_t)(kt0 + kr) * HEAD_D + ks * 8,
                    kbase + kr * (KSTRIDE * 2) + ks * 16);
    async_copy_b128(Kh + (size_t)(kt0 + kr + 16) * HEAD_D + ks * 8,
                    kbase + (kr + 16) * (KSTRIDE * 2) + ks * 16);
    // V^T chunk: 64 rows x 64B; thread -> (row, row+32) x one 16B segment
    const int vr = tid >> 2, vs = tid & 3;
    async_copy_b128(VTh + (size_t)vr * T_SEQ + kt0 + vs * 8,
                    vbase + vr * (VSTRIDE * 2) + vs * 16);
    async_copy_b128(VTh + (size_t)(vr + 32) * T_SEQ + kt0 + vs * 8,
                    vbase + (vr + 32) * (VSTRIDE * 2) + vs * 16);
  };

  const int nchunks = q0b / 32 + 2;    // covers keys up to q0b + 63
  stage(0, 0);

  for (int c = 0; c < nchunks; ++c) {
    const int cur = c & 1;
    wait_asynccnt0();                  // own async transfers complete
    __syncthreads();                   // everyone's transfers visible; prev reads done
    if (c + 1 < nchunks) stage(c + 1, cur ^ 1);

    const int kt0 = c * 32;
    if (kt0 > q0 + 15) continue;       // beyond this wave's causal bound

    const bf16* lk = lds_k[cur];
    const bf16* lv = lds_v[cur];

    // ---- hoist all fragment loads: ds_loads clause up, waits overlap ----
    v16bf bk00 = load_frag(lk + 0,                 KSTRIDE);
    v16bf bk01 = load_frag(lk + 32,                KSTRIDE);
    v16bf bk10 = load_frag(lk + 16 * KSTRIDE,      KSTRIDE);
    v16bf bk11 = load_frag(lk + 16 * KSTRIDE + 32, KSTRIDE);
    v16bf bv0  = load_frag(lv +  0 * VSTRIDE, VSTRIDE);
    v16bf bv1  = load_frag(lv + 16 * VSTRIDE, VSTRIDE);
    v16bf bv2  = load_frag(lv + 32 * VSTRIDE, VSTRIDE);
    v16bf bv3  = load_frag(lv + 48 * VSTRIDE, VSTRIDE);

    // ---- S = Q * K^T for 32 keys: two 16x16 f32 tiles (4 WMMA) ----
    v8f s0 = {}, s1 = {};
    s0 = wmma_bf16(aq0, bk00, s0);
    s0 = wmma_bf16(aq1, bk01, s0);
    s1 = wmma_bf16(aq0, bk10, s1);
    s1 = wmma_bf16(aq1, bk11, s1);

    // ---- C-layout -> LDS (f32, padded rows) ----
#pragma unroll
    for (int i = 0; i < 8; ++i) {
      const int m = i + 8 * half;
      ls[m * SSTRIDE + n]      = s0[i];
      ls[m * SSTRIDE + 16 + n] = s1[i];
    }

    // ---- re-read this lane's A-layout slice: row n, k = kb+0..7, kb+16..23 ----
    const float* rowp = ls + n * SSTRIDE + kb;
    float4 va = *(const float4*)(rowp);
    float4 vb = *(const float4*)(rowp + 4);
    float4 vc = *(const float4*)(rowp + 16);
    float4 vd = *(const float4*)(rowp + 20);
    float sv[16] = { va.x, va.y, va.z, va.w, vb.x, vb.y, vb.z, vb.w,
                     vc.x, vc.y, vc.z, vc.w, vd.x, vd.y, vd.z, vd.w };

    // ---- causal mask only on partial chunks (wave-uniform branch) ----
    if (kt0 + 31 > q0) {
      const int lim = q0 + n - kt0;    // largest allowed (k - kt0) for this row
#pragma unroll
      for (int j = 0; j < 8; ++j) {
        if (kb + j > lim)      sv[j]     = -1e30f;
        if (kb + 16 + j > lim) sv[j + 8] = -1e30f;
      }
    }

    // ---- online softmax: row owned by lanes {n, n+16} ----
    float mx = sv[0];
#pragma unroll
    for (int j = 1; j < 16; ++j) mx = fmaxf(mx, sv[j]);
    mx = fmaxf(mx, __shfl_xor(mx, 16, 32));          // combine the two k-halves

    const float mnew    = fmaxf(m_run, mx);
    const float alpha_r = exp2f(m_run - mnew);
    m_run = mnew;

    v16bf ap;                                        // P directly in A layout
    float rs = 0.0f;
#pragma unroll
    for (int j = 0; j < 16; ++j) {
      const float p = exp2f(sv[j] - mnew);
      rs += p;
      ap[j] = (bf16)p;
    }
    rs += __shfl_xor(rs, 16, 32);
    l_run = l_run * alpha_r + rs;

    // ---- rescale accumulators: broadcast alpha from row-owner lanes ----
#pragma unroll
    for (int i = 0; i < 8; ++i) {
      const float a = __shfl(alpha_r, i + 8 * half, 32);
      acc0[i] *= a; acc1[i] *= a; acc2[i] *= a; acc3[i] *= a;
    }

    // ---- O += P * V ----
    acc0 = wmma_bf16(ap, bv0, acc0);
    acc1 = wmma_bf16(ap, bv1, acc1);
    acc2 = wmma_bf16(ap, bv2, acc2);
    acc3 = wmma_bf16(ap, bv3, acc3);
  }

  // ---- normalize and store to pre-projection activation (bf16) ----
  const int b = bh >> 4, h = bh & 15;
#pragma unroll
  for (int i = 0; i < 8; ++i) {
    const int m = i + 8 * half;
    const float li   = __shfl(l_run, i + 8 * half, 32);
    const float invl = 1.0f / li;
    const size_t row = ((size_t)b * T_SEQ + q0 + m) * C_EMB + h * HEAD_D;
    Aout[row +  0 + n] = (bf16)(acc0[i] * invl);
    Aout[row + 16 + n] = (bf16)(acc1[i] * invl);
    Aout[row + 32 + n] = (bf16)(acc2[i] * invl);
    Aout[row + 48 + n] = (bf16)(acc3[i] * invl);
  }
}

// ---------------------------------------------------------------------------
// Host side
// ---------------------------------------------------------------------------
extern "C" void kernel_launch(void* const* d_in, const int* in_sizes, int n_in,
                              void* d_out, int out_size, void* d_ws, size_t ws_size,
                              hipStream_t stream) {
  (void)in_sizes; (void)n_in; (void)out_size; (void)ws_size;

  const float* x  = (const float*)d_in[0];
  const float* Wq = (const float*)d_in[1];
  const float* Wk = (const float*)d_in[2];
  const float* Wv = (const float*)d_in[3];
  const float* Wo = (const float*)d_in[4];
  float* out = (float*)d_out;

  // Workspace carve-up (48 MiB of bf16)
  const size_t NX = (size_t)M_ROWS * C_EMB;   // 4,194,304
  const size_t NW = (size_t)C_EMB * C_EMB;    // 1,048,576
  bf16* xb   = (bf16*)d_ws;
  bf16* Wqb  = xb  + NX;
  bf16* Wkb  = Wqb + NW;
  bf16* Wvb  = Wkb + NW;
  bf16* Wob  = Wvb + NW;
  bf16* Qb   = Wob + NW;
  bf16* Kb   = Qb  + NX;
  bf16* VTb  = Kb  + NX;
  bf16* Aout = VTb + NX;

  // 1) converts
  cvt_f32_bf16<<<dim3((unsigned)((NX + 255) / 256)), dim3(256), 0, stream>>>(x,  xb,  (int)NX);
  cvt_f32_bf16<<<dim3((unsigned)((NW + 255) / 256)), dim3(256), 0, stream>>>(Wq, Wqb, (int)NW);
  cvt_f32_bf16<<<dim3((unsigned)((NW + 255) / 256)), dim3(256), 0, stream>>>(Wk, Wkb, (int)NW);
  cvt_f32_bf16<<<dim3((unsigned)((NW + 255) / 256)), dim3(256), 0, stream>>>(Wv, Wvb, (int)NW);
  cvt_f32_bf16<<<dim3((unsigned)((NW + 255) / 256)), dim3(256), 0, stream>>>(Wo, Wob, (int)NW);

  // 2) Q/K projections: M=4096 (tokens), N=1024 (channels)
  dim3 gblk(128);
  dim3 ggrid(M_ROWS / 128, C_EMB / 64);
  gemm_xwT<0><<<ggrid, gblk, 0, stream>>>(xb, Wqb, Qb, nullptr);
  gemm_xwT<1><<<ggrid, gblk, 0, stream>>>(xb, Wkb, Kb, nullptr);
  //    V^T via swapped GEMM: M=1024 (channels), N=4096 (tokens)
  dim3 vgrid(C_EMB / 128, M_ROWS / 64);
  gemm_xwT<2><<<vgrid, gblk, 0, stream>>>(Wvb, xb, VTb, nullptr);

  // 3) flash attention: grid = (T/64 query blocks, B*H heads)
  flash_attn<<<dim3(T_SEQ / 64, 2 * N_HEAD), dim3(128), 0, stream>>>(Qb, Kb, VTb, Aout);

  // 4) output projection, fp32 result
  gemm_xwT<3><<<ggrid, gblk, 0, stream>>>(Aout, Wob, nullptr, out);
}
